// SimpleAttention_4578435138114
// MI455X (gfx1250) — matrix operand order repaired and checked by reading, not compiled, not run on hardware
//
#include <hip/hip_runtime.h>
#include <hip/hip_bf16.h>

typedef __attribute__((ext_vector_type(16))) _Float16 v16h;
typedef __attribute__((ext_vector_type(8)))  _Float16 v8h;
typedef __attribute__((ext_vector_type(8)))  float    v8f;

// Problem constants (match reference setup_inputs)
#define BB   16
#define CC   3072
#define TT   2048
#define NVAR 32      // i dimension
#define DEP  96      // channels per variable
#define JJ   64      // softmax/contract dim (32 from x + 32 from x_instant)
#define DD   64      // context channels per variable
#define TTILE 16     // t values per workgroup

// LDS layout (dynamic, 192 KB total):
//   [0      , 128KB) : phase 1-2: f32 logits  [i][j][t]  (32*64*16*4)
//                      phase 3-4: f16 ctx     [t][j][d]  (16*64*64*2)  (aliased)
//   [128KB  , 192KB) : f16 probs [t][i][j]               (16*32*64*2)
#define SMEM_BYTES (196608)
#define SA_OFF     (NVAR * JJ * TTILE * 4)   // 131072

__global__ __launch_bounds__(256)
void simple_attention_wmma(const float* __restrict__ x,
                           const float* __restrict__ xi,
                           float* __restrict__ out,
                           float* __restrict__ att)
{
    extern __shared__ char smem[];
    float*    sL = (float*)smem;                 // logits  [i][j][t]
    _Float16* sB = (_Float16*)smem;              // ctx f16 [t][j][d]   (aliases sL)
    _Float16* sA = (_Float16*)(smem + SA_OFF);   // probs   [t][i][j]

    const int tid = threadIdx.x;
    const int b   = blockIdx.y;
    const int t0  = blockIdx.x * TTILE;

    const float* xb  = x  + (size_t)b * CC * TT;
    const float* xib = xi + (size_t)b * CC * TT;

    // ---------------- Phase 1: load attention logits (coalesced 64B rows) ----
    // 2048 rows (i in [0,32), j in [0,64)), 16 floats each.
    #pragma unroll
    for (int r = 0; r < 8; ++r) {
        const int ridx = r * 256 + tid;          // == i*64 + j
        const int i = ridx >> 6;
        const int j = ridx & 63;
        const float* src = (j < NVAR) ? xb : xib;
        const int c = i * DEP + (j & 31);
        const float4* g = (const float4*)(src + (size_t)c * TT + t0);
        float4 v0 = g[0], v1 = g[1], v2 = g[2], v3 = g[3];
        // warm L2 with this variable's context rows for phase 3 (gfx1250 global_prefetch)
        __builtin_prefetch(src + (size_t)(i * DEP + NVAR + (j & 31)) * TT + t0, 0, 1);
        float4* dst = (float4*)(sL + (size_t)ridx * TTILE);
        dst[0] = v0; dst[1] = v1; dst[2] = v2; dst[3] = v3;
    }
    __syncthreads();

    // ---------------- Phase 2: softmax over j (64) per (i,t) ------------------
    // 512 rows -> 2 per thread. Emits f16 probs to sA and the `attentions`
    // output (a1 + off-diagonal a2) directly, 64B-coalesced over t.
    #pragma unroll
    for (int r = 0; r < 2; ++r) {
        const int idx = r * 256 + tid;           // tid 0..15 -> same i, t=0..15
        const int i = idx >> 4;
        const int t = idx & 15;
        const float* row = sL + (size_t)i * JJ * TTILE + t;  // stride TTILE per j
        float v[JJ];
        float m = -1e30f;
        #pragma unroll
        for (int j = 0; j < JJ; ++j) { v[j] = row[(size_t)j * TTILE]; m = fmaxf(m, v[j]); }
        float s = 0.0f;
        #pragma unroll
        for (int j = 0; j < JJ; ++j) { v[j] = __expf(v[j] - m); s += v[j]; }
        const float inv = 1.0f / s;
        _Float16* pa = sA + ((size_t)t * NVAR + i) * JJ;
        #pragma unroll
        for (int j = 0; j < JJ; ++j) pa[j] = (_Float16)(v[j] * inv);
        float* ao = att + (((size_t)b * NVAR + i) * NVAR) * TT + (t0 + t);
        #pragma unroll
        for (int j = 0; j < NVAR; ++j) {
            const float p2 = (i == j) ? 0.0f : v[j + NVAR] * inv;
            ao[(size_t)j * TT] = v[j] * inv + p2;
        }
    }
    __syncthreads();  // all reads of sL done before aliasing as sB

    // ---------------- Phase 3: load context, convert f32 -> f16 ---------------
    // 64 j-rows x 32 d-pairs = 2048 work items; each loads two 64B rows and
    // scatters 16 packed half2 into sB[t][j][d].
    #pragma unroll
    for (int r = 0; r < 8; ++r) {
        const int ridx = r * 256 + tid;
        const int j  = ridx >> 5;
        const int dp = ridx & 31;                // d = 2*dp
        const float* src = (j < NVAR) ? xb : xib;
        const int c = (j & 31) * DEP + NVAR + dp * 2;
        const float4* g0 = (const float4*)(src + (size_t)c * TT + t0);
        const float4* g1 = (const float4*)(src + (size_t)(c + 1) * TT + t0);
        float a0[TTILE], a1[TTILE];
        ((float4*)a0)[0] = g0[0]; ((float4*)a0)[1] = g0[1];
        ((float4*)a0)[2] = g0[2]; ((float4*)a0)[3] = g0[3];
        ((float4*)a1)[0] = g1[0]; ((float4*)a1)[1] = g1[1];
        ((float4*)a1)[2] = g1[2]; ((float4*)a1)[3] = g1[3];
        #pragma unroll
        for (int t = 0; t < TTILE; ++t) {
            union { _Float16 h[2]; unsigned int u; } p;
            p.h[0] = (_Float16)a0[t];
            p.h[1] = (_Float16)a1[t];
            *(unsigned int*)(sB + (((size_t)t * JJ + j) * DD + dp * 2)) = p.u;
        }
    }
    __syncthreads();

    // ---------------- Phase 4: WMMA  P(32x64) @ Ctx(64x64) per t --------------
    // 8 waves: wave w -> output tile (itile = w>>2, dtile = w&3).
    const int lane  = tid & 31;
    const int w     = tid >> 5;
    const int itile = w >> 2;                    // 0..1
    const int dtile = w & 3;                     // 0..3
    const int m     = lane & 15;
    const int hi    = lane >> 4;                 // 0/1
    const int kb    = hi * 8;                    // A-layout K base inside 32-chunk
    float* outB = out + (size_t)b * 2048 * TT + t0;

    for (int t = 0; t < TTILE; ++t) {
        const _Float16* aRow = sA + ((size_t)t * NVAR + itile * 16 + m) * JJ;
        v8f acc = {};
        #pragma unroll
        for (int kk = 0; kk < 2; ++kk) {
            // A fragment (16-bit A 16x32 layout): lanes 0-15 K={0..7,16..23},
            // lanes 16-31 K={8..15,24..31}; two contiguous 16B LDS chunks.
            v8h lo = *(const v8h*)(aRow + kk * 32 + kb);
            v8h hv = *(const v8h*)(aRow + kk * 32 + kb + 16);
            v16h aFrag = __builtin_shufflevector(lo, hv,
                0,1,2,3,4,5,6,7,8,9,10,11,12,13,14,15);
            // B fragment: lane holds column N = dtile*16 + m, K in same order as A
            // (K={0..7,16..23} / {8..15,24..31}); strided LDS gather.
            v16h bFrag;
            #pragma unroll
            for (int s = 0; s < 16; ++s) {
                const int k = kk * 32 + kb + (s < 8 ? s : s + 8);
                bFrag[s] = sB[((size_t)t * JJ + k) * DD + dtile * 16 + m];
            }
            acc = __builtin_amdgcn_wmma_f32_16x16x32_f16(
                false, aFrag, false, bFrag, (short)0, acc, false, false);
        }
        // D layout: VGPR r, lanes 0-15 -> M=r, N=lane; lanes 16-31 -> M=r+8.
        #pragma unroll
        for (int r8 = 0; r8 < 8; ++r8) {
            const int ig = itile * 16 + r8 + hi * 8;
            const int dg = dtile * 16 + m;
            outB[(size_t)(ig * DD + dg) * TT + t] = acc[r8];
        }
    }
}

extern "C" void kernel_launch(void* const* d_in, const int* in_sizes, int n_in,
                              void* d_out, int out_size, void* d_ws, size_t ws_size,
                              hipStream_t stream) {
    const float* x  = (const float*)d_in[0];
    const float* xi = (const float*)d_in[1];
    float* out = (float*)d_out;
    float* att = out + (size_t)BB * 2048 * TT;   // 16*2048*2048 floats, then attentions

    hipFuncSetAttribute((const void*)simple_attention_wmma,
                        hipFuncAttributeMaxDynamicSharedMemorySize, SMEM_BYTES);

    dim3 grid(TT / TTILE, BB);                   // (128, 16)
    simple_attention_wmma<<<grid, 256, SMEM_BYTES, stream>>>(x, xi, out, att);
}
